// GCN_75625784148347
// MI455X (gfx1250) — matrix operand order repaired and checked by reading, not compiled, not run on hardware
//
#include <hip/hip_runtime.h>
#include <hip/hip_bf16.h>
#include <math.h>

// ---------------- problem constants ----------------
#define NN   100000
#define EE   800000
#define F_IN 64
#define HID  128

typedef _Float16 h16;
typedef __attribute__((ext_vector_type(16))) _Float16 v16h;
typedef __attribute__((ext_vector_type(8)))  _Float16 v8h;
typedef __attribute__((ext_vector_type(8)))  float    v8f;

// Hardware f32 atomic add, no return value -> lowers to global_atomic_add_f32
// (no-return form, STOREcnt-tracked) instead of a CAS loop.
__device__ __forceinline__ void atomic_add_f32(float* p, float v) {
    (void)__hip_atomic_fetch_add(p, v, __ATOMIC_RELAXED, __HIP_MEMORY_SCOPE_AGENT);
}

// ---------------- WMMA helpers ----------------
__device__ __forceinline__ v8f wmma_f16(v16h a, v16h b, v8f c) {
    // D = A(16x32 f16) x B(32x16 f16) + C(16x16 f32)
    return __builtin_amdgcn_wmma_f32_16x16x32_f16(
        /*neg_a=*/false, a, /*neg_b=*/false, b,
        /*c_mod=*/(short)0, c, /*reuse_a=*/false, /*reuse_b=*/false);
}

// A-matrix fragment from a row-major f32 activation row.
// Lane (group g = lane/16) holds K chunks [kt*32 + g*8, +8) and [kt*32 + 16 + g*8, +8).
__device__ __forceinline__ v16h a_frag_f32(const float* __restrict__ row, int kt, int g) {
    const float* p0 = row + kt * 32 + g * 8;
    const float* p1 = p0 + 16;
    v16h a;
#pragma unroll
    for (int i = 0; i < 8; ++i) {
        a[i]     = (h16)p0[i];
        a[8 + i] = (h16)p1[i];
    }
    return a;
}

// A-matrix fragment from a row-major f16 activation row.
__device__ __forceinline__ v16h a_frag_f16(const h16* __restrict__ row, int kt, int g) {
    v8h lo = *(const v8h*)(row + kt * 32 + g * 8);
    v8h hi = *(const v8h*)(row + kt * 32 + 16 + g * 8);
    v16h a;
#pragma unroll
    for (int i = 0; i < 8; ++i) {
        a[i]     = lo[i];
        a[8 + i] = hi[i];
    }
    return a;
}

// B-matrix fragment: lane holds K chunks [g*8, +8) and [16 + g*8, +8) of column n.
// B[k][n] = W[n][k]  (row-major W, K-contiguous) -> straight vector load.
__device__ __forceinline__ v16h b_frag_f16(const h16* __restrict__ wrow, int kt, int g) {
    v8h lo = *(const v8h*)(wrow + kt * 32 + g * 8);
    v8h hi = *(const v8h*)(wrow + kt * 32 + 16 + g * 8);
    v16h b;
#pragma unroll
    for (int i = 0; i < 8; ++i) {
        b[i]     = lo[i];
        b[8 + i] = hi[i];
    }
    return b;
}

// ---------------- small utility kernels ----------------
__global__ void __launch_bounds__(256) cvt_f32_to_f16(const float* __restrict__ src,
                                                      h16* __restrict__ dst, int count) {
    int i = blockIdx.x * blockDim.x + threadIdx.x;
    if (i < count) dst[i] = (h16)src[i];
}

__global__ void __launch_bounds__(256) zero_f32(float* __restrict__ p, long long count4) {
    long long i = (long long)blockIdx.x * blockDim.x + threadIdx.x;
    if (i < count4) ((float4*)p)[i] = make_float4(0.f, 0.f, 0.f, 0.f);
}

// ---------------- edge scatter: agg1[dst] += x[src], 64 f32 feats ----------------
__global__ void __launch_bounds__(256) scatter_f32_64(const float* __restrict__ x,
                                                      const long long* __restrict__ ei,
                                                      float* __restrict__ agg) {
    long long t = (long long)blockIdx.x * blockDim.x + threadIdx.x;
    if (t >= (long long)EE * 16) return;
    int e = (int)(t >> 4);
    int f = (int)(t & 15) * 4;
    long long s = ei[e];
    long long d = ei[EE + e];
    float4 v = *(const float4*)(x + s * F_IN + f);
    float* ap = agg + d * F_IN + f;
    atomic_add_f32(ap + 0, v.x);
    atomic_add_f32(ap + 1, v.y);
    atomic_add_f32(ap + 2, v.z);
    atomic_add_f32(ap + 3, v.w);
}

// ---------------- edge scatter: agg2[dst] += h1[src], 128 f16 feats -> f32 acc ----------------
__global__ void __launch_bounds__(256) scatter_f16_128(const h16* __restrict__ h1,
                                                       const long long* __restrict__ ei,
                                                       float* __restrict__ agg) {
    long long t = (long long)blockIdx.x * blockDim.x + threadIdx.x;
    if (t >= (long long)EE * 16) return;
    int e = (int)(t >> 4);
    int f = (int)(t & 15) * 8;
    long long s = ei[e];
    long long d = ei[EE + e];
    v8h v = *(const v8h*)(h1 + s * HID + f);
    float* ap = agg + d * HID + f;
#pragma unroll
    for (int i = 0; i < 8; ++i) atomic_add_f32(ap + i, (float)v[i]);
}

// ---------------- layer 1: h1 = elu(agg @ W_rel.T + b + x @ W_root.T), f16 out ----------------
// wave = 16 rows x 128 cols; K = 64 (2 k-slices) per matrix -> 4 WMMAs per 16x16 col tile.
__global__ void __launch_bounds__(256) layer1_wmma(const float* __restrict__ agg,
                                                   const float* __restrict__ x,
                                                   const h16* __restrict__ w_rel,
                                                   const float* __restrict__ bias,
                                                   const h16* __restrict__ w_root,
                                                   h16* __restrict__ h1) {
    int wave = threadIdx.x >> 5;
    int lane = threadIdx.x & 31;
    int row_base = (blockIdx.x * 8 + wave) * 16;
    if (row_base >= NN) return;           // wave-uniform: EXEC stays all-ones for WMMA
    int g  = lane >> 4;
    int ln = lane & 15;

    const float* arow = agg + (size_t)(row_base + ln) * F_IN;
    const float* xrow = x   + (size_t)(row_base + ln) * F_IN;
    v16h A0 = a_frag_f32(arow, 0, g), A1 = a_frag_f32(arow, 1, g);
    v16h X0 = a_frag_f32(xrow, 0, g), X1 = a_frag_f32(xrow, 1, g);

#pragma unroll
    for (int ct = 0; ct < 8; ++ct) {
        int col = ct * 16 + ln;
        const h16* wr = w_rel  + (size_t)col * F_IN;
        const h16* wo = w_root + (size_t)col * F_IN;
        v16h Br0 = b_frag_f16(wr, 0, g), Br1 = b_frag_f16(wr, 1, g);
        v16h Bo0 = b_frag_f16(wo, 0, g), Bo1 = b_frag_f16(wo, 1, g);
        v8f c = {};
        c = wmma_f16(A0, Br0, c);
        c = wmma_f16(A1, Br1, c);
        c = wmma_f16(X0, Bo0, c);
        c = wmma_f16(X1, Bo1, c);
        float b = bias[col];
#pragma unroll
        for (int j = 0; j < 8; ++j) {
            float v = c[j] + b;
            v = (v > 0.f) ? v : (__expf(v) - 1.f);   // ELU
            h1[(size_t)(row_base + j + 8 * g) * HID + col] = (h16)v;
        }
    }
}

// ---------------- layer 2: h2 = agg2 @ W_rel.T + b + h1 @ W_root.T, f32 out ----------------
// K = 128 (4 k-slices) per matrix -> 8 WMMAs per 16x16 col tile.
__global__ void __launch_bounds__(256) layer2_wmma(const float* __restrict__ agg,
                                                   const h16* __restrict__ h1,
                                                   const h16* __restrict__ w_rel,
                                                   const float* __restrict__ bias,
                                                   const h16* __restrict__ w_root,
                                                   float* __restrict__ h2) {
    int wave = threadIdx.x >> 5;
    int lane = threadIdx.x & 31;
    int row_base = (blockIdx.x * 8 + wave) * 16;
    if (row_base >= NN) return;
    int g  = lane >> 4;
    int ln = lane & 15;

    const float* arow = agg + (size_t)(row_base + ln) * HID;
    const h16*   rrow = h1  + (size_t)(row_base + ln) * HID;
    v16h A[4], R[4];
#pragma unroll
    for (int kt = 0; kt < 4; ++kt) {
        A[kt] = a_frag_f32(arow, kt, g);
        R[kt] = a_frag_f16(rrow, kt, g);
    }

#pragma unroll
    for (int ct = 0; ct < 8; ++ct) {
        int col = ct * 16 + ln;
        const h16* wr = w_rel  + (size_t)col * HID;
        const h16* wo = w_root + (size_t)col * HID;
        v8f c = {};
#pragma unroll
        for (int kt = 0; kt < 4; ++kt) c = wmma_f16(A[kt], b_frag_f16(wr, kt, g), c);
#pragma unroll
        for (int kt = 0; kt < 4; ++kt) c = wmma_f16(R[kt], b_frag_f16(wo, kt, g), c);
        float b = bias[col];
#pragma unroll
        for (int j = 0; j < 8; ++j)
            h2[(size_t)(row_base + j + 8 * g) * HID + col] = c[j] + b;
    }
}

// ---------------- final MLP: out = relu(h2 @ Wfc1.T + b1) @ Wfc2.T + b2 ----------------
__global__ void __launch_bounds__(256) fc_kernel(const float* __restrict__ h2,
                                                 const float* __restrict__ Wfc1,
                                                 const float* __restrict__ bfc1,
                                                 const float* __restrict__ Wfc2,
                                                 const float* __restrict__ bfc2,
                                                 float* __restrict__ out) {
    int n = blockIdx.x * blockDim.x + threadIdx.x;
    if (n >= NN) return;
    const float* row = h2 + (size_t)n * HID;
    float acc2 = bfc2[0];
#pragma unroll 4
    for (int o = 0; o < 20; ++o) {
        const float* w = Wfc1 + o * HID;
        float a = bfc1[o];
#pragma unroll
        for (int k = 0; k < HID; k += 4) {
            float4 r = *(const float4*)(row + k);
            float4 ww = *(const float4*)(w + k);
            a = fmaf(r.x, ww.x, a);
            a = fmaf(r.y, ww.y, a);
            a = fmaf(r.z, ww.z, a);
            a = fmaf(r.w, ww.w, a);
        }
        acc2 = fmaf(fmaxf(a, 0.f), Wfc2[o], acc2);
    }
    out[n] = acc2;
}

// ---------------- host launcher ----------------
extern "C" void kernel_launch(void* const* d_in, const int* in_sizes, int n_in,
                              void* d_out, int out_size, void* d_ws, size_t ws_size,
                              hipStream_t stream) {
    const float*     x       = (const float*)d_in[0];
    const long long* ei      = (const long long*)d_in[1];
    const float*     W1_rel  = (const float*)d_in[2];
    const float*     b1      = (const float*)d_in[3];
    const float*     W1_root = (const float*)d_in[4];
    const float*     W2_rel  = (const float*)d_in[5];
    const float*     b2      = (const float*)d_in[6];
    const float*     W2_root = (const float*)d_in[7];
    const float*     W_fc1   = (const float*)d_in[8];
    const float*     b_fc1   = (const float*)d_in[9];
    const float*     W_fc2   = (const float*)d_in[10];
    const float*     b_fc2   = (const float*)d_in[11];
    float* out = (float*)d_out;

    char* ws = (char*)d_ws;
    // workspace layout (bytes), all 256-aligned
    float* agg1 = (float*)(ws + 0);                       // N*64*4  = 25,600,000
    float* agg2 = (float*)(ws + 25600000);                // N*128*4 = 51,200,000
    h16*   h1   = (h16*)  (ws + 76800000);                // N*128*2 = 25,600,000
    float* h2   = (float*)(ws + 102400000);               // N*128*4 = 51,200,000
    h16*   w1r  = (h16*)  (ws + 153600000);               // 8192 halves
    h16*   w1o  = (h16*)  (ws + 153616384);               // 8192 halves
    h16*   w2r  = (h16*)  (ws + 153632768);               // 16384 halves
    h16*   w2o  = (h16*)  (ws + 153665536);               // 16384 halves

    // 1) weight conversion f32 -> f16
    cvt_f32_to_f16<<<(HID * F_IN + 255) / 256, 256, 0, stream>>>(W1_rel,  w1r, HID * F_IN);
    cvt_f32_to_f16<<<(HID * F_IN + 255) / 256, 256, 0, stream>>>(W1_root, w1o, HID * F_IN);
    cvt_f32_to_f16<<<(HID * HID + 255) / 256, 256, 0, stream>>>(W2_rel,  w2r, HID * HID);
    cvt_f32_to_f16<<<(HID * HID + 255) / 256, 256, 0, stream>>>(W2_root, w2o, HID * HID);

    // 2) zero agg1 + agg2 (contiguous region: N*(64+128) floats)
    {
        long long count4 = (long long)NN * (F_IN + HID) / 4;
        long long blocks = (count4 + 255) / 256;
        zero_f32<<<(int)blocks, 256, 0, stream>>>(agg1, count4);
    }

    // 3) layer-1 edge aggregation
    {
        long long work = (long long)EE * 16;
        scatter_f32_64<<<(int)((work + 255) / 256), 256, 0, stream>>>(x, ei, agg1);
    }

    // 4) layer-1 GEMM (WMMA) + bias + ELU -> h1 (f16)
    {
        int row_tiles = NN / 16;                    // 6250
        int blocks = (row_tiles + 7) / 8;           // 8 waves/block
        layer1_wmma<<<blocks, 256, 0, stream>>>(agg1, x, w1r, b1, w1o, h1);
    }

    // 5) layer-2 edge aggregation (gather f16 h1, f32 atomic acc)
    {
        long long work = (long long)EE * 16;
        scatter_f16_128<<<(int)((work + 255) / 256), 256, 0, stream>>>(h1, ei, agg2);
    }

    // 6) layer-2 GEMM (WMMA) + bias -> h2 (f32)
    {
        int row_tiles = NN / 16;
        int blocks = (row_tiles + 7) / 8;
        layer2_wmma<<<blocks, 256, 0, stream>>>(agg2, h1, w2r, b2, w2o, h2);
    }

    // 7) final MLP -> out [N,1]
    fc_kernel<<<(NN + 255) / 256, 256, 0, stream>>>(h2, W_fc1, b_fc1, W_fc2, b_fc2, out);
}
